// AAGCN_27608049778953
// MI455X (gfx1250) — compile-verified, matrix operand order checked by reading, not run on hardware
//
#include <hip/hip_runtime.h>
#include <hip/hip_bf16.h>

// ---------------------------------------------------------------------------
// AAGCN forward on gfx1250 (MI455X).
// Heavy channel GEMMs -> v_wmma_f32_16x16x32_bf16 (bf16 inputs, f32 accum).
// 9-tap temporal conv folded into the GEMM K-loop (tap-outer, register accum).
// Staging: uniform in-bounds fast path (no guards at all for interior tiles),
// multiply-masked slow path (no EXEC branches), pair-packed b32 DS stores,
// aligned b128 DS fragment loads. Small V=17 mixes / attention / BN -> VALU.
// ---------------------------------------------------------------------------

typedef __bf16 bf16_t;
typedef bf16_t v16bf __attribute__((ext_vector_type(16)));
typedef bf16_t v8bf  __attribute__((ext_vector_type(8)));
typedef float  v8f   __attribute__((ext_vector_type(8)));

#define T_DIM  300
#define V_DIM  17
#define TVC    (T_DIM * V_DIM)   // 5100
#define NB     16                // N*M
#define EPSV   1e-5f

__device__ __forceinline__ bf16_t to_bf16(float f) { return (bf16_t)f; }

__device__ __forceinline__ float block_reduce_256(float v) {
  __shared__ float red[256];
  int t = threadIdx.x;
  red[t] = v; __syncthreads();
  for (int s = 128; s > 0; s >>= 1) {
    if (t < s) red[t] += red[t + s];
    __syncthreads();
  }
  float r = red[0]; __syncthreads();
  return r;
}

// ---------------------------------------------------------------------------
// WMMA GEMM with fused temporal taps:
//   Y[n] (O x Ncol) = sum_{tap} Wtap (O x K) * shift_tap(X[n]) (K x Ncol) [+bias]
// W element (o,k,tap) at W[o*wRow + k*wCol + tap]; pointwise = (K,1,taps=1).
// Tap 'tap' shifts B columns by (tap - taps/2)*V_DIM with zero padding.
// Block = 128 threads (4 waves); tile 64x64; each wave owns a 32x32 quad.
// LDS: lA [m][k], lB [n][k] (both k-contiguous -> b128 fragment loads).
// ---------------------------------------------------------------------------
__global__ __launch_bounds__(128)
void gemm_bf16_wmma(const float* __restrict__ W, const float* __restrict__ Xp,
                    const float* __restrict__ bias, float* __restrict__ Yp,
                    int O, int K, int Ncol, int wRow, int wCol, int taps,
                    long strideX, long strideY) {
  __shared__ bf16_t lA[64 * 32];   // [m][k]
  __shared__ bf16_t lB[64 * 32];   // [n][k] (transposed stage)
  unsigned* lA32 = (unsigned*)lA;
  unsigned* lB32 = (unsigned*)lB;

  const int nb = blockIdx.z;
  const float* X = Xp + (long)nb * strideX;
  float*       Y = Yp + (long)nb * strideY;
  const int m0 = blockIdx.y * 64, n0 = blockIdx.x * 64;
  const int tid = threadIdx.x, lane = tid & 31, w = tid >> 5;
  const int wm = (w >> 1) * 32, wn = (w & 1) * 32;
  const int laneN = lane & 15;
  const int kHi = (lane >> 4) << 3;   // lanes 16..31 hold K+8 halves (ISA 7.12.2)
  const bool fullM = (m0 + 64) <= O;  // wave-uniform
  v8f acc[2][2] = {};

  for (int tap = 0; tap < taps; ++tap) {
    const int shift = (tap - (taps >> 1)) * V_DIM;
    const bool fullCol = (n0 + shift) >= 0 && (n0 + 64 + shift) <= Ncol;  // uniform
    for (int k0 = 0; k0 < K; k0 += 32) {
      const bool fullK = (k0 + 32) <= K;  // uniform

      // ---- Stage A 64x32, pair-packed b32 stores (conflict-free)
      if (fullM && fullK) {
#pragma unroll
        for (int i = 0; i < 8; ++i) {
          int q = tid + i * 128;        // 0..1023 pair slots
          int kp = q & 15, m = q >> 4;  // consecutive tid -> consecutive k-pair
          long base = (long)(m0 + m) * wRow + (long)(k0 + 2 * kp) * wCol + tap;
          float v0 = W[base];
          float v1 = W[base + wCol];
          union { bf16_t h[2]; unsigned u; } pk;
          pk.h[0] = to_bf16(v0); pk.h[1] = to_bf16(v1);
          lA32[m * 16 + kp] = pk.u;
        }
      } else {
#pragma unroll
        for (int i = 0; i < 8; ++i) {
          int q = tid + i * 128;
          int kp = q & 15, m = q >> 4;
          int k = 2 * kp;
          int mc  = min(m0 + m, O - 1);
          int kc0 = min(k0 + k, K - 1);
          int kc1 = min(k0 + k + 1, K - 1);
          float v0 = W[(long)mc * wRow + (long)kc0 * wCol + tap];
          float v1 = W[(long)mc * wRow + (long)kc1 * wCol + tap];
          bool mok = (m0 + m) < O;
          // multiply-mask: load unconditionally consumed -> no EXEC branches
          v0 *= (mok && (k0 + k)     < K) ? 1.f : 0.f;
          v1 *= (mok && (k0 + k + 1) < K) ? 1.f : 0.f;
          union { bf16_t h[2]; unsigned u; } pk;
          pk.h[0] = to_bf16(v0); pk.h[1] = to_bf16(v1);
          lA32[m * 16 + kp] = pk.u;
        }
      }

      // ---- Stage B 32x64 -> LDS [n][k], coalesced over columns
      if (fullCol && fullK) {
#pragma unroll
        for (int i = 0; i < 8; ++i) {
          int q = tid + i * 128;
          int n = q & 63, kp = q >> 6;  // consecutive tid -> consecutive column
          long base = (long)(k0 + 2 * kp) * Ncol + (n0 + n + shift);
          float v0 = X[base];
          float v1 = X[base + Ncol];
          union { bf16_t h[2]; unsigned u; } pk;
          pk.h[0] = to_bf16(v0); pk.h[1] = to_bf16(v1);
          lB32[n * 16 + kp] = pk.u;
        }
      } else {
#pragma unroll
        for (int i = 0; i < 8; ++i) {
          int q = tid + i * 128;
          int n = q & 63, kp = q >> 6;
          int k = 2 * kp;
          int col  = n0 + n + shift;
          int colc = min(max(col, 0), Ncol - 1);
          int kc0 = min(k0 + k, K - 1);
          int kc1 = min(k0 + k + 1, K - 1);
          float v0 = X[(long)kc0 * Ncol + colc];
          float v1 = X[(long)kc1 * Ncol + colc];
          bool cok = (col >= 0) && (col < Ncol);
          v0 *= (cok && (k0 + k)     < K) ? 1.f : 0.f;
          v1 *= (cok && (k0 + k + 1) < K) ? 1.f : 0.f;
          union { bf16_t h[2]; unsigned u; } pk;
          pk.h[0] = to_bf16(v0); pk.h[1] = to_bf16(v1);
          lB32[n * 16 + kp] = pk.u;
        }
      }
      __syncthreads();

      // ---- Fragments: two aligned 16B LDS vector loads per fragment
      auto frag = [&](const bf16_t* base, int row) -> v16bf {
        v8bf lo = *(const v8bf*)(base + row * 32 + kHi);
        v8bf hi = *(const v8bf*)(base + row * 32 + kHi + 16);
        return __builtin_shufflevector(lo, hi, 0, 1, 2, 3, 4, 5, 6, 7,
                                       8, 9, 10, 11, 12, 13, 14, 15);
      };
      v16bf a0 = frag(lA, wm + laneN);
      v16bf a1 = frag(lA, wm + 16 + laneN);
      v16bf b0 = frag(lB, wn + laneN);
      v16bf b1 = frag(lB, wn + 16 + laneN);

      acc[0][0] = __builtin_amdgcn_wmma_f32_16x16x32_bf16(
          false, a0, false, b0, (short)0, acc[0][0], false, false);
      acc[0][1] = __builtin_amdgcn_wmma_f32_16x16x32_bf16(
          false, a0, false, b1, (short)0, acc[0][1], false, false);
      acc[1][0] = __builtin_amdgcn_wmma_f32_16x16x32_bf16(
          false, a1, false, b0, (short)0, acc[1][0], false, false);
      acc[1][1] = __builtin_amdgcn_wmma_f32_16x16x32_bf16(
          false, a1, false, b1, (short)0, acc[1][1], false, false);
      __syncthreads();
    }
  }

  const int rowHi = (lane >> 4) << 3;  // C/D layout: VGPR r -> M=r (+8 hi lanes)
#pragma unroll
  for (int i = 0; i < 2; ++i)
#pragma unroll
    for (int j = 0; j < 2; ++j)
#pragma unroll
      for (int r = 0; r < 8; ++r) {
        int M  = m0 + wm + 16 * i + r + rowHi;
        int Nc = n0 + wn + 16 * j + laneN;
        if (M < O && Nc < Ncol) {
          float v = acc[i][j][r];
          if (bias) v += bias[M];
          Y[(long)M * Ncol + Nc] = v;
        }
      }
}

// ---------------------------------------------------------------------------
// data_bn: x[8,2,300,17,3] -> h0[16,3,300,17], BN over channel (m*V+v)*3+c
// ---------------------------------------------------------------------------
__global__ void data_bn_kernel(const float* __restrict__ x, const float* g,
                               const float* b, const float* m, const float* v,
                               float* __restrict__ out, long total) {
  long idx = (long)blockIdx.x * blockDim.x + threadIdx.x;
  if (idx >= total) return;
  int vv = (int)(idx % V_DIM); long r = idx / V_DIM;
  int t = (int)(r % T_DIM); r /= T_DIM;
  int c = (int)(r % 3); int nm = (int)(r / 3);
  float val = x[(((long)nm * T_DIM + t) * V_DIM + vv) * 3 + c];
  int ch = ((nm & 1) * V_DIM + vv) * 3 + c;
  out[idx] = (val - m[ch]) * rsqrtf(v[ch] + EPSV) * g[ch] + b[ch];
}

// Ai[n,v,w] = A[v,w] + alpha * tanh( sum_{c,t} a1[n,c,t,v]*a2[n,c,t,w] / (inter*T) )
__global__ void att_kernel(const float* __restrict__ a1, const float* __restrict__ a2,
                           const float* __restrict__ Asub, const float* __restrict__ alpha,
                           float* __restrict__ Ai, int inter) {
  int n = blockIdx.x, vw = blockIdx.y;
  int v = vw / V_DIM, w = vw % V_DIM;
  int CT = inter * T_DIM;
  float s = 0.f;
  for (int i = threadIdx.x; i < CT; i += blockDim.x) {
    int c = i / T_DIM, t = i % T_DIM;
    long base = ((long)(n * inter + c) * T_DIM + t) * V_DIM;
    s += a1[base + v] * a2[base + w];
  }
  float tot = block_reduce_256(s);
  if (threadIdx.x == 0)
    Ai[(n * V_DIM + v) * V_DIM + w] =
        Asub[v * V_DIM + w] + alpha[0] * tanhf(tot / (float)CT);
}

// y[n,o,t,w] (beta? += : =) sum_v g[n,o,t,v]*Ai[n,v,w] + bd[o]
__global__ void vmix_kernel(const float* __restrict__ g, const float* __restrict__ Ai,
                            const float* __restrict__ bd, float* __restrict__ y,
                            int C, int beta, long total) {
  long idx = (long)blockIdx.x * blockDim.x + threadIdx.x;
  if (idx >= total) return;
  int w = (int)(idx % V_DIM);
  long ct = idx / V_DIM;
  int t = (int)(ct % T_DIM);
  long nc = ct / T_DIM;
  int n = (int)(nc / C), o = (int)(nc % C);
  const float* gp = g + (nc * T_DIM + t) * V_DIM;
  const float* Ap = Ai + (long)n * V_DIM * V_DIM;
  float s = bd[o];
#pragma unroll
  for (int v = 0; v < V_DIM; ++v) s += gp[v] * Ap[v * V_DIM + w];
  if (beta) y[idx] += s; else y[idx] = s;
}

// out = relu( bn(y) + res? (rg? bn2(res) : res) : 0 )
__global__ void bn_add_relu_kernel(const float* __restrict__ y,
    const float* g, const float* b, const float* m, const float* vv,
    const float* __restrict__ res,
    const float* rg, const float* rb, const float* rm, const float* rv,
    float* __restrict__ out, int C, long total) {
  long idx = (long)blockIdx.x * blockDim.x + threadIdx.x;
  if (idx >= total) return;
  int c = (int)((idx / (long)TVC) % C);
  float val = (y[idx] - m[c]) * rsqrtf(vv[c] + EPSV) * g[c] + b[c];
  if (res) {
    float r = res[idx];
    if (rg) r = (r - rm[c]) * rsqrtf(rv[c] + EPSV) * rg[c] + rb[c];
    val += r;
  }
  out[idx] = fmaxf(val, 0.f);
}

// means
__global__ void mean_t_kernel(const float* __restrict__ y, float* __restrict__ o, int C) {
  int idx = blockIdx.x * blockDim.x + threadIdx.x;
  if (idx >= NB * C * V_DIM) return;
  int v = idx % V_DIM; int nc = idx / V_DIM;
  const float* p = y + (long)nc * TVC + v;
  float s = 0.f;
  for (int t = 0; t < T_DIM; ++t) s += p[t * V_DIM];
  o[idx] = s * (1.f / T_DIM);
}
__global__ void mean_v_kernel(const float* __restrict__ y, float* __restrict__ o, int C) {
  int idx = blockIdx.x * blockDim.x + threadIdx.x;
  if (idx >= NB * C * T_DIM) return;
  int t = idx % T_DIM; int nc = idx / T_DIM;
  const float* p = y + ((long)nc * T_DIM + t) * V_DIM;
  float s = 0.f;
#pragma unroll
  for (int v = 0; v < V_DIM; ++v) s += p[v];
  o[idx] = s * (1.f / V_DIM);
}
__global__ void mean_tv_kernel(const float* __restrict__ y, float* __restrict__ o, int C) {
  int n = blockIdx.x, c = blockIdx.y;
  const float* p = y + (long)(n * C + c) * TVC;
  float s = 0.f;
  for (int i = threadIdx.x; i < TVC; i += blockDim.x) s += p[i];
  float tot = block_reduce_256(s);
  if (threadIdx.x == 0) o[n * C + c] = tot * (1.f / TVC);
}

// spatial attention: s[n,v] = sigmoid( sum_{c,j} wsa[c,j]*mv[n,c,v+j-8] + bsa )
__global__ void satt_kernel(const float* __restrict__ mv, const float* __restrict__ wsa,
                            const float* __restrict__ bsa, float* __restrict__ s, int C) {
  int n = blockIdx.x, v = blockIdx.y;
  float acc = 0.f;
  for (int i = threadIdx.x; i < C * V_DIM; i += blockDim.x) {
    int c = i / V_DIM, j = i % V_DIM;
    int vv = v + j - (V_DIM - 1) / 2;
    if (vv >= 0 && vv < V_DIM) acc += wsa[c * V_DIM + j] * mv[(n * C + c) * V_DIM + vv];
  }
  float tot = block_reduce_256(acc);
  if (threadIdx.x == 0) s[n * V_DIM + v] = 1.f / (1.f + expf(-(tot + bsa[0])));
}
// temporal attention: s[n,t] = sigmoid( sum_{c,j} wta[c,j]*mt[n,c,t+j-4] + bta )
__global__ void tatt_kernel(const float* __restrict__ mt, const float* __restrict__ wta,
                            const float* __restrict__ bta, float* __restrict__ s, int C) {
  int n = blockIdx.x, t = blockIdx.y;
  float acc = 0.f;
  for (int i = threadIdx.x; i < C * 9; i += blockDim.x) {
    int c = i / 9, j = i % 9;
    int tt = t + j - 4;
    if (tt >= 0 && tt < T_DIM) acc += wta[c * 9 + j] * mt[(n * C + c) * T_DIM + tt];
  }
  float tot = block_reduce_256(acc);
  if (threadIdx.x == 0) s[n * T_DIM + t] = 1.f / (1.f + expf(-(tot + bta[0])));
}

__global__ void scale_v_kernel(float* __restrict__ y, const float* __restrict__ s,
                               int C, long total) {
  long idx = (long)blockIdx.x * blockDim.x + threadIdx.x;
  if (idx >= total) return;
  int v = (int)(idx % V_DIM);
  int n = (int)(idx / ((long)C * TVC));
  y[idx] *= 1.f + s[n * V_DIM + v];
}
__global__ void scale_t_kernel(float* __restrict__ y, const float* __restrict__ s,
                               int C, long total) {
  long idx = (long)blockIdx.x * blockDim.x + threadIdx.x;
  if (idx >= total) return;
  int t = (int)((idx / V_DIM) % T_DIM);
  int n = (int)(idx / ((long)C * TVC));
  y[idx] *= 1.f + s[n * T_DIM + t];
}
__global__ void scale_c_kernel(float* __restrict__ y, const float* __restrict__ s,
                               int C, long total) {
  long idx = (long)blockIdx.x * blockDim.x + threadIdx.x;
  if (idx >= total) return;
  int c = (int)((idx / (long)TVC) % C);
  int n = (int)(idx / ((long)C * TVC));
  y[idx] *= 1.f + s[n * C + c];
}

// SE FCs
__global__ void fc1_kernel(const float* __restrict__ m, const float* __restrict__ w1,
                           const float* __restrict__ b1, float* __restrict__ h, int C, int C2) {
  int idx = blockIdx.x * blockDim.x + threadIdx.x;
  if (idx >= NB * C2) return;
  int n = idx / C2, k = idx % C2;
  float s = b1[k];
  for (int c = 0; c < C; ++c) s += w1[k * C + c] * m[n * C + c];
  h[idx] = fmaxf(s, 0.f);
}
__global__ void fc2_kernel(const float* __restrict__ h, const float* __restrict__ w2,
                           const float* __restrict__ b2, float* __restrict__ s, int C2, int C) {
  int idx = blockIdx.x * blockDim.x + threadIdx.x;
  if (idx >= NB * C) return;
  int n = idx / C, c = idx % C;
  float acc = b2[c];
  for (int k = 0; k < C2; ++k) acc += w2[c * C2 + k] * h[n * C2 + k];
  s[idx] = 1.f / (1.f + expf(-acc));
}

// ---------------------------------------------------------------------------
// Host orchestration
// ---------------------------------------------------------------------------
namespace {
struct Cursor {
  void* const* in; int i;
  const float* f() { return (const float*)in[i++]; }
};
struct BP {
  const float *A, *alpha, *b1, *b2, *ba, *bb, *bd, *bn_b, *bn_g, *bn_m, *bn_v,
      *bsa, *bta, *down_b, *down_bn_b, *down_bn_g, *down_bn_m, *down_bn_v, *down_w,
      *w1, *w2, *wa, *wb, *wd, *wsa, *wta,
      *res_b, *res_bn_b, *res_bn_g, *res_bn_m, *res_bn_v, *res_w,
      *tcn_b, *tcn_bn_b, *tcn_bn_g, *tcn_bn_m, *tcn_bn_v, *tcn_w;
};
static const int CFG_CI[10] = {3, 64, 64, 64, 64, 128, 128, 128, 256, 256};
static const int CFG_CO[10] = {64, 64, 64, 64, 128, 128, 128, 256, 256, 256};
}  // namespace

extern "C" void kernel_launch(void* const* d_in, const int* in_sizes, int n_in,
                              void* d_out, int out_size, void* d_ws, size_t ws_size,
                              hipStream_t stream) {
  (void)in_sizes; (void)n_in; (void)out_size; (void)ws_size;
  const float* x = (const float*)d_in[0];
  Cursor cur{d_in, 1};

  // JAX pytree flatten order: blocks (list, each dict key-sorted) then data_bn.
  BP bp[10];
  for (int i = 0; i < 10; ++i) {
    int ci = CFG_CI[i], co = CFG_CO[i];
    bool down = (ci != co), res = (i > 0) && (ci != co);
    BP& p = bp[i];
    p.A = cur.f(); p.alpha = cur.f(); p.b1 = cur.f(); p.b2 = cur.f();
    p.ba = cur.f(); p.bb = cur.f(); p.bd = cur.f();
    p.bn_b = cur.f(); p.bn_g = cur.f(); p.bn_m = cur.f(); p.bn_v = cur.f();
    p.bsa = cur.f(); p.bta = cur.f();
    if (down) {
      p.down_b = cur.f(); p.down_bn_b = cur.f(); p.down_bn_g = cur.f();
      p.down_bn_m = cur.f(); p.down_bn_v = cur.f(); p.down_w = cur.f();
    } else { p.down_w = nullptr; p.down_b = p.down_bn_b = p.down_bn_g = p.down_bn_m = p.down_bn_v = nullptr; }
    p.w1 = cur.f(); p.w2 = cur.f(); p.wa = cur.f(); p.wb = cur.f(); p.wd = cur.f();
    p.wsa = cur.f(); p.wta = cur.f();
    if (res) {
      p.res_b = cur.f(); p.res_bn_b = cur.f(); p.res_bn_g = cur.f();
      p.res_bn_m = cur.f(); p.res_bn_v = cur.f(); p.res_w = cur.f();
    } else { p.res_w = nullptr; p.res_b = p.res_bn_b = p.res_bn_g = p.res_bn_m = p.res_bn_v = nullptr; }
    p.tcn_b = cur.f(); p.tcn_bn_b = cur.f(); p.tcn_bn_g = cur.f();
    p.tcn_bn_m = cur.f(); p.tcn_bn_v = cur.f(); p.tcn_w = cur.f();
  }
  const float* dbn_b = cur.f(); const float* dbn_g = cur.f();
  const float* dbn_m = cur.f(); const float* dbn_v = cur.f();

  // Workspace partition (floats)
  float* ws = (float*)d_ws;
  const long BIG = 16L * 256 * TVC;
  const long AMAX = 16L * 64 * TVC;
  float* H0 = ws;  ws += BIG;
  float* Hp = ws;  ws += BIG;
  float* Yb = ws;  ws += BIG;   // gcn accumulator
  float* Gb = ws;  ws += BIG;   // WdX scratch / down-res / tcn output
  float* A1 = ws;  ws += AMAX;
  float* A2 = ws;  ws += AMAX;
  float* Aib = ws; ws += 16 * V_DIM * V_DIM;
  float* MV = ws;  ws += 16L * 256 * V_DIM;
  float* MT = ws;  ws += 16L * 256 * T_DIM;
  float* Sv = ws;  ws += 16 * T_DIM;
  float* Mc = ws;  ws += 16 * 256;
  float* Hse = ws; ws += 16 * 128;
  float* Sc = ws;  ws += 16 * 256;

  auto gemm = [&](const float* W, const float* X, const float* bias, float* Y,
                  int O, int K, int wRow, int wCol, int taps) {
    dim3 g((TVC + 63) / 64, (O + 63) / 64, 16);
    gemm_bf16_wmma<<<g, 128, 0, stream>>>(W, X, bias, Y, O, K, TVC, wRow, wCol,
                                          taps, (long)K * TVC, (long)O * TVC);
  };

  // data_bn + reshape -> H0 [16,3,300,17]
  {
    long tot = 16L * 3 * TVC;
    data_bn_kernel<<<(unsigned)((tot + 255) / 256), 256, 0, stream>>>(
        x, dbn_g, dbn_b, dbn_m, dbn_v, H0, tot);
  }

  float* hin = H0; float* hout = Hp;
  for (int i = 0; i < 10; ++i) {
    const int ci = CFG_CI[i], co = CFG_CO[i], inter = co / 4;
    const BP& p = bp[i];
    const long totCo = 16L * co * TVC;
    const unsigned gEl = (unsigned)((totCo + 255) / 256);

    // ---- unit_aagcn: y = sum_s (Wd_s X) Ai_s + bd_s ----
    for (int s = 0; s < 3; ++s) {
      gemm(p.wa + (long)s * inter * ci, hin, p.ba + s * inter, A1, inter, ci, ci, 1, 1);
      gemm(p.wb + (long)s * inter * ci, hin, p.bb + s * inter, A2, inter, ci, ci, 1, 1);
      att_kernel<<<dim3(16, V_DIM * V_DIM), 256, 0, stream>>>(
          A1, A2, p.A + s * V_DIM * V_DIM, p.alpha, Aib, inter);
      gemm(p.wd + (long)s * co * ci, hin, nullptr, Gb, co, ci, ci, 1, 1);
      vmix_kernel<<<gEl, 256, 0, stream>>>(Gb, Aib, p.bd + s * co, Yb, co, s > 0 ? 1 : 0, totCo);
    }
    // BN + residual(down or identity) + ReLU, in place on Yb
    if (p.down_w) {
      gemm(p.down_w, hin, p.down_b, Gb, co, ci, ci, 1, 1);
      bn_add_relu_kernel<<<gEl, 256, 0, stream>>>(
          Yb, p.bn_g, p.bn_b, p.bn_m, p.bn_v,
          Gb, p.down_bn_g, p.down_bn_b, p.down_bn_m, p.down_bn_v, Yb, co, totCo);
    } else {
      bn_add_relu_kernel<<<gEl, 256, 0, stream>>>(
          Yb, p.bn_g, p.bn_b, p.bn_m, p.bn_v,
          hin, nullptr, nullptr, nullptr, nullptr, Yb, co, totCo);
    }
    // spatial attention
    mean_t_kernel<<<(NB * co * V_DIM + 255) / 256, 256, 0, stream>>>(Yb, MV, co);
    satt_kernel<<<dim3(16, V_DIM), 256, 0, stream>>>(MV, p.wsa, p.bsa, Sv, co);
    scale_v_kernel<<<gEl, 256, 0, stream>>>(Yb, Sv, co, totCo);
    // temporal attention
    mean_v_kernel<<<(NB * co * T_DIM + 255) / 256, 256, 0, stream>>>(Yb, MT, co);
    tatt_kernel<<<dim3(16, T_DIM), 256, 0, stream>>>(MT, p.wta, p.bta, Sv, co);
    scale_t_kernel<<<gEl, 256, 0, stream>>>(Yb, Sv, co, totCo);
    // channel SE
    mean_tv_kernel<<<dim3(16, co), 256, 0, stream>>>(Yb, Mc, co);
    fc1_kernel<<<(NB * (co / 2) + 255) / 256, 256, 0, stream>>>(Mc, p.w1, p.b1, Hse, co, co / 2);
    fc2_kernel<<<(NB * co + 255) / 256, 256, 0, stream>>>(Hse, p.w2, p.b2, Sc, co / 2, co);
    scale_c_kernel<<<gEl, 256, 0, stream>>>(Yb, Sc, co, totCo);

    // ---- unit_tcn (9x1): single launch, taps folded into the K-loop ----
    gemm(p.tcn_w, Yb, p.tcn_b, Gb, co, co, co * 9, 9, 9);

    // BN + block residual + ReLU -> next h (or d_out on last block)
    float* dst = (i == 9) ? (float*)d_out : hout;
    if (i == 0) {
      bn_add_relu_kernel<<<gEl, 256, 0, stream>>>(
          Gb, p.tcn_bn_g, p.tcn_bn_b, p.tcn_bn_m, p.tcn_bn_v,
          nullptr, nullptr, nullptr, nullptr, nullptr, dst, co, totCo);
    } else if (ci == co) {
      bn_add_relu_kernel<<<gEl, 256, 0, stream>>>(
          Gb, p.tcn_bn_g, p.tcn_bn_b, p.tcn_bn_m, p.tcn_bn_v,
          hin, nullptr, nullptr, nullptr, nullptr, dst, co, totCo);
    } else {
      gemm(p.res_w, hin, p.res_b, dst, co, ci, ci, 1, 1);
      bn_add_relu_kernel<<<gEl, 256, 0, stream>>>(
          Gb, p.tcn_bn_g, p.tcn_bn_b, p.tcn_bn_m, p.tcn_bn_v,
          dst, p.res_bn_g, p.res_bn_b, p.res_bn_m, p.res_bn_v, dst, co, totCo);
    }
    float* tmp = hin; hin = hout; hout = tmp;
  }
}
